// PixelSegmenter_66606352826630
// MI455X (gfx1250) — matrix operand order repaired and checked by reading, not compiled, not run on hardware
//
#include <hip/hip_runtime.h>
#include <hip/hip_bf16.h>

typedef __attribute__((ext_vector_type(16))) _Float16 v16h;
typedef __attribute__((ext_vector_type(8)))  _Float16 v8h;
typedef __attribute__((ext_vector_type(4)))  _Float16 v4h;
typedef __attribute__((ext_vector_type(8)))  float    v8f;

union AFrag {
    v16h v;
    v8h  o[2];
    v4h  q[4];
};

// LDS layout. Zero-padded regions are contiguous at the front so one memset
// loop covers them. Alignment: every WMMA fragment load below is >=8B aligned
// (row pitches: xl 256B, h1 416B, w1l 320B, w2l 960B; all multiples of 32B
// except w1l which is a multiple of 32B too: 160*2=320).
struct __align__(16) Smem {
    _Float16 xl[16][28][128];   // x as f16, [b][y][px*4+ic], pad ic=3, px>=28 -> 0
    _Float16 h1[16][12][208];   // layer1 out, [b][y][px*16+c], 13-px pitch, px=12 -> 0
    _Float16 w1l[16][160];      // [oc][ky*32 + kx*4 + ic], pads zero
    _Float16 w2l[32][480];      // [oc][ky*96 + kx*16 + c], kx=5 pad zero
    _Float16 h2[16][512];       // layer2 out, [b][(y*4+x)*32 + c]
    _Float16 w3l[512];          // [(y*4+x)*32 + c]
    float magred[16][2][8];
    float aggred[16][8];
    float magv[16][2];
    float aggv[16];
};

#define ZERO_BYTES (sizeof(_Float16)*(16*28*128 + 16*12*208 + 16*160 + 32*480))

__global__ __launch_bounds__(256) void pixel_segmenter_kernel(
    const float* __restrict__ xg,
    const float* __restrict__ w1g, const float* __restrict__ b1g,
    const float* __restrict__ w2g, const float* __restrict__ b2g,
    const float* __restrict__ w3g, const float* __restrict__ b3g,
    const float* __restrict__ wmagg, const float* __restrict__ bmagg,
    const float* __restrict__ wfg, const float* __restrict__ bfg,
    float* __restrict__ outg)
{
    __shared__ Smem sm;
    const int t     = threadIdx.x;
    const int lane  = t & 31;
    const int wv    = t >> 5;      // wave 0..7 (wave32)
    const int h     = lane >> 4;   // half-lane
    const int bm    = lane & 15;   // M-row (batch) for A, N-col (channel) for B/C
    const int bbase = blockIdx.x * 16;

    // ---- phase 0: zero the padded LDS regions ----
    {
        uint32_t* p = (uint32_t*)&sm;
        const int nz = (int)(ZERO_BYTES / 4);
        for (int i = t; i < nz; i += 256) p[i] = 0u;
    }
    __syncthreads();

    // ---- phase 1: stage inputs/weights into LDS (f32 -> f16) ----
    for (int i = t; i < 16 * 2352; i += 256) {              // x: coalesced
        int b = i / 2352, e = i % 2352;
        int ic = e / 784, r = e % 784;
        int y = r / 28, xx = r % 28;
        float v = xg[(size_t)(bbase + b) * 2352 + e];
        sm.xl[b][y][xx * 4 + ic] = (_Float16)v;
    }
    for (int i = t; i < 16 * 75; i += 256) {                // w1 [oc][ic][ky][kx]
        int oc = i / 75, r = i % 75;
        int ic = r / 25, r2 = r % 25, ky = r2 / 5, kx = r2 % 5;
        sm.w1l[oc][ky * 32 + kx * 4 + ic] = (_Float16)w1g[i];
    }
    for (int i = t; i < 32 * 400; i += 256) {               // w2 [oc][ic][ky][kx]
        int oc = i / 400, r = i % 400;
        int ic = r / 25, r2 = r % 25, ky = r2 / 5, kx = r2 % 5;
        sm.w2l[oc][ky * 96 + kx * 16 + ic] = (_Float16)w2g[i];
    }
    for (int i = t; i < 512; i += 256) {                    // w3 [0][c][y][x]
        int c = i / 16, r = i % 16;                         // r = y*4+x
        sm.w3l[r * 32 + c] = (_Float16)w3g[i];
    }
    __syncthreads();

    // ---- phase 2: layer1 conv(3->16,5x5)+bias+relu+pool2 via WMMA ----
    // 144 pooled positions, wave-uniform split (18 per wave), EXEC all-ones.
    {
        const float bias1 = b1g[bm];
        for (int pp = wv; pp < 144; pp += 8) {
            int py = pp / 12, px = pp % 12;
            v8f m; for (int v = 0; v < 8; ++v) m[v] = -3.0e38f;
            for (int d = 0; d < 4; ++d) {
                int oy = 2 * py + (d >> 1), ox = 2 * px + (d & 1);
                v8f c = {};
                for (int ky = 0; ky < 5; ++ky) {
                    const _Float16* arow = &sm.xl[bm][oy + ky][0];
                    AFrag a, w;
                    // A lane layout: e0..7 = K 8h..8h+7, e8..15 = K 8h+16..8h+23
                    // K = px_local*4 + ic  -> 8 halves == 2 pixels, contiguous
                    int p0 = ox + 2 * h;
                    a.q[0] = *(const v4h*)&arow[p0 * 4];
                    a.q[1] = *(const v4h*)&arow[p0 * 4 + 4];
                    a.q[2] = *(const v4h*)&arow[(p0 + 4) * 4];
                    a.q[3] = *(const v4h*)&arow[(p0 + 4) * 4 + 4];
                    // B lane layout: N = bm, e0..7 = K 8h..8h+7, e8..15 = K 8h+16..8h+23
                    const _Float16* brow = &sm.w1l[bm][ky * 32 + 8 * h];
                    w.o[0] = *(const v8h*)&brow[0];
                    w.o[1] = *(const v8h*)&brow[16];
                    c = __builtin_amdgcn_wmma_f32_16x16x32_f16(
                            false, a.v, false, w.v, (short)0, c, false, false);
                }
                for (int v = 0; v < 8; ++v) m[v] = fmaxf(m[v], c[v]);
            }
            // C layout: lane -> channel N=bm, VGPR v -> batch M = v + 8h
            for (int v = 0; v < 8; ++v) {
                float f = fmaxf(m[v] + bias1, 0.0f);
                sm.h1[v + 8 * h][py][px * 16 + bm] = (_Float16)f;
            }
        }
    }
    __syncthreads();

    // ---- phase 3: layer2 conv(16->32,5x5)+bias+relu+pool2 via WMMA ----
    // 16 pooled positions x 2 N-tiles = 32 units, 4 per wave.
    for (int u = wv * 4; u < wv * 4 + 4; ++u) {
        int pp = u >> 1, nt = u & 1;
        int py = pp >> 2, px = pp & 3;
        const float bias2 = b2g[nt * 16 + bm];
        v8f m; for (int v = 0; v < 8; ++v) m[v] = -3.0e38f;
        for (int d = 0; d < 4; ++d) {
            int oy = 2 * py + (d >> 1), ox = 2 * px + (d & 1);
            v8f c = {};
            for (int ky = 0; ky < 5; ++ky) {
                const _Float16* arow  = &sm.h1[bm][oy + ky][0];
                const _Float16* brow0 = &sm.w2l[nt * 16 + bm][ky * 96 + 8 * h];
                for (int sub = 0; sub < 3; ++sub) {
                    AFrag a, w;
                    // chunk covers pixels ox+2sub, ox+2sub+1; 8 halves = half a
                    // pixel's channels (8h..8h+7), 16B aligned -> ds_load_b128
                    int pix = ox + 2 * sub;
                    a.o[0] = *(const v8h*)&arow[pix * 16 + 8 * h];
                    a.o[1] = *(const v8h*)&arow[(pix + 1) * 16 + 8 * h];
                    const _Float16* brow = brow0 + sub * 32;
                    w.o[0] = *(const v8h*)&brow[0];
                    w.o[1] = *(const v8h*)&brow[16];
                    c = __builtin_amdgcn_wmma_f32_16x16x32_f16(
                            false, a.v, false, w.v, (short)0, c, false, false);
                }
            }
            for (int v = 0; v < 8; ++v) m[v] = fmaxf(m[v], c[v]);
        }
        for (int v = 0; v < 8; ++v) {
            float f = fmaxf(m[v] + bias2, 0.0f);
            sm.h2[v + 8 * h][(py * 4 + px) * 32 + nt * 16 + bm] = (_Float16)f;
        }
    }
    __syncthreads();

    // ---- phase 4: partial dots (deterministic tree reductions, no atomics) ----
    {   // mag: all 256 threads, t = sl*32 + ch*16 + b, 2352 = 8*294
        int b = t & 15, ch = (t >> 4) & 1, sl = t >> 5;
        float s = 0.f;
        const float* wm = wmagg + ch * 2352;
        for (int j = 0; j < 294; ++j) {
            int e = sl * 294 + j;
            int ic = e / 784, r = e % 784;
            int y = r / 28, xx = r % 28;
            s += (float)sm.xl[b][y][xx * 4 + ic] * wm[e];
        }
        sm.magred[b][ch][sl] = s;
    }
    if (t < 128) {      // layer3: b = t/8, 512 = 8*64
        int b = t >> 3, sl = t & 7;
        float s = 0.f;
        for (int j = 0; j < 64; ++j) {
            int k = sl * 64 + j;
            s += (float)sm.h2[b][k] * (float)sm.w3l[k];
        }
        sm.aggred[b][sl] = s;
    }
    __syncthreads();
    if (t < 32) {
        int b = t & 15, ch = t >> 4;
        float s = 0.f;
        for (int i = 0; i < 8; ++i) s += sm.magred[b][ch][i];
        sm.magv[b][ch] = fmaxf(s + bmagg[ch], 0.f);
    } else if (t < 48) {
        int b = t - 32;
        float s = 0.f;
        for (int i = 0; i < 8; ++i) s += sm.aggred[b][i];
        sm.aggv[b] = fmaxf(s + b3g[0], 0.f);
    }
    __syncthreads();

    // ---- phase 5: fusion 1x1 conv + relu, write output ----
    if (t < 16) {
        float o = wfg[0] * sm.aggv[t] + wfg[1] * sm.magv[t][0]
                + wfg[2] * sm.magv[t][1] + bfg[0];
        outg[bbase + t] = fmaxf(o, 0.f);
    }
}

extern "C" void kernel_launch(void* const* d_in, const int* in_sizes, int n_in,
                              void* d_out, int out_size, void* d_ws, size_t ws_size,
                              hipStream_t stream) {
    (void)n_in; (void)d_ws; (void)ws_size; (void)out_size;
    const float* x    = (const float*)d_in[0];
    const float* w1   = (const float*)d_in[1];
    const float* b1   = (const float*)d_in[2];
    const float* w2   = (const float*)d_in[3];
    const float* b2   = (const float*)d_in[4];
    const float* w3   = (const float*)d_in[5];
    const float* b3   = (const float*)d_in[6];
    const float* wmag = (const float*)d_in[7];
    const float* bmag = (const float*)d_in[8];
    const float* wf   = (const float*)d_in[9];
    const float* bf   = (const float*)d_in[10];
    float* out = (float*)d_out;

    int B = in_sizes[0] / (3 * 28 * 28);   // 8192
    int nblk = B / 16;                     // 512 workgroups of 16 samples
    pixel_segmenter_kernel<<<nblk, 256, 0, stream>>>(
        x, w1, b1, w2, b2, w3, b3, wmag, bmag, wf, bf, out);
}
